// TransformerBlock_45801531245143
// MI455X (gfx1250) — compile-verified, matrix operand order checked by reading, not run on hardware
//
#include <hip/hip_runtime.h>
#include <math.h>
#include <stdint.h>

// ---------------------------------------------------------------------------
// Types
// ---------------------------------------------------------------------------
typedef __bf16 bf16;
typedef __bf16 v16bf __attribute__((ext_vector_type(16)));
typedef __bf16 v8bf  __attribute__((ext_vector_type(8)));
typedef float  v8f   __attribute__((ext_vector_type(8)));

#define BB 8
#define SS 1024
#define DD 1024
#define HH 16
#define DH 64
#define FF 4096
#define NTOK (BB * SS)          // 8192

// ---------------------------------------------------------------------------
// WMMA helpers (CDNA5 v_wmma_f32_16x16x32_bf16)
// ---------------------------------------------------------------------------
__device__ __forceinline__ v8f wmma_bf16(v16bf a, v16bf b, v8f c) {
  return __builtin_amdgcn_wmma_f32_16x16x32_bf16(
      /*neg_a=*/false, a, /*neg_b=*/false, b,
      /*c_mod=*/(short)0, c, /*reuse_a=*/false, /*reuse_b=*/false);
}

// A/B fragment loader. ISA 16-bit 16x32 layout: lane half h holds
// K[8h..8h+7] in halves 0..7 and K[16+8h..16+8h+7] in halves 8..15.
// Caller passes p = &row[k0 + 8*half]; we read p[0..7] and p[16..23].
__device__ __forceinline__ v16bf frag_from(const bf16* p) {
  v8bf lo = *(const v8bf*)(p);
  v8bf hi = *(const v8bf*)(p + 16);
  v16bf a;
#pragma unroll
  for (int i = 0; i < 8; ++i) { a[i] = lo[i]; a[i + 8] = hi[i]; }
  return a;
}

// Strided B fragment (column-gather from a row-major LDS tile).
__device__ __forceinline__ v16bf frag_strided(const bf16* base, int stride, int half) {
  v16bf a;
#pragma unroll
  for (int i = 0; i < 8; ++i) {
    a[i]     = base[(8 * half + i) * stride];
    a[i + 8] = base[(16 + 8 * half + i) * stride];
  }
  return a;
}

// CDNA5 async global->LDS copy of one 16B chunk (tracked by ASYNCcnt; bypasses
// VGPRs entirely).  GV addressing: vdst = LDS byte address, v[addr64], off.
__device__ __forceinline__ void async_copy16(const bf16* lds_dst, const bf16* gsrc) {
  uint32_t lds_off = (uint32_t)(uint64_t)lds_dst;  // flat LDS addr[31:0] = LDS offset
  uint64_t ga = (uint64_t)gsrc;
  asm volatile("global_load_async_to_lds_b128 %0, %1, off"
               :: "v"(lds_off), "v"(ga) : "memory");
}

__device__ __forceinline__ void wait_async0() {
  asm volatile("s_wait_asynccnt 0x0" ::: "memory");
}

// ---------------------------------------------------------------------------
// fp32 -> bf16 conversion (grid-stride)
// ---------------------------------------------------------------------------
__global__ void cvt_f32_bf16(const float* __restrict__ in, bf16* __restrict__ out,
                             size_t n) {
  size_t i = (size_t)blockIdx.x * blockDim.x + threadIdx.x;
  size_t stride = (size_t)gridDim.x * blockDim.x;
  for (; i < n; i += stride) out[i] = (bf16)in[i];
}

// ---------------------------------------------------------------------------
// Tiled bf16 WMMA GEMM:  Y[M,N] = act( (A[M,K] * W[N,K]^T + bias) * scale )
// A, W row-major with K contiguous (torch Linear layout -> NT GEMM).
// Block tile 128x128, 8 waves of 32x64, K-step 32, double-buffered LDS filled
// by async global->LDS DMA.
// ---------------------------------------------------------------------------
template <bool GELU, bool OUT_BF16>
__global__ __launch_bounds__(256)
void gemm_bf16_nt(const bf16* __restrict__ A, const bf16* __restrict__ W,
                  const float* __restrict__ bias, float scale,
                  float* outF, bf16* outB, int M, int N, int K) {
  __shared__ bf16 As[2][128][40];   // 40-half row stride keeps 16B chunk align
  __shared__ bf16 Bs[2][128][40];

  const int tid  = threadIdx.x;
  const int lane = tid & 31;
  const int wave = tid >> 5;
  const int wm   = wave & 3;       // 4 M-bands of 32
  const int wn   = wave >> 2;      // 2 N-bands of 64
  const int half = lane >> 4;
  const int l16  = lane & 15;
  const int m0   = blockIdx.x * 128;
  const int n0   = blockIdx.y * 128;

  v8f acc[2][4] = {};

  // 128x32 tile = 512 16B chunks; 256 threads -> 2 chunks each (A and B).
  auto load_tile = [&](int buf, int k0) {
#pragma unroll
    for (int c = 0; c < 2; ++c) {
      int ch = tid + c * 256;                  // 0..511
      int r = ch >> 2, col = (ch & 3) * 8;
      async_copy16(&As[buf][r][col], &A[(size_t)(m0 + r) * K + k0 + col]);
      async_copy16(&Bs[buf][r][col], &W[(size_t)(n0 + r) * K + k0 + col]);
    }
  };

  load_tile(0, 0);
  wait_async0();
  __syncthreads();

  const int KT = K >> 5;
  for (int kt = 0; kt < KT; ++kt) {
    const int cur = kt & 1;
    if (kt + 1 < KT) load_tile(cur ^ 1, (kt + 1) << 5);
    if (kt + 2 < KT) {  // L2 warm-up of the k+2 slab (global_prefetch_b8)
      __builtin_prefetch(&A[(size_t)(m0 + (tid >> 1)) * K + ((kt + 2) << 5)], 0, 1);
      __builtin_prefetch(&W[(size_t)(n0 + (tid >> 1)) * K + ((kt + 2) << 5)], 0, 1);
    }

    v16bf a0 = frag_from(&As[cur][wm * 32 + l16][8 * half]);
    v16bf a1 = frag_from(&As[cur][wm * 32 + 16 + l16][8 * half]);
#pragma unroll
    for (int ni = 0; ni < 4; ++ni) {
      v16bf b = frag_from(&Bs[cur][wn * 64 + ni * 16 + l16][8 * half]);
      acc[0][ni] = wmma_bf16(a0, b, acc[0][ni]);
      acc[1][ni] = wmma_bf16(a1, b, acc[1][ni]);
    }

    wait_async0();       // next-buffer DMA landed (ASYNCcnt not covered by barrier)
    __syncthreads();     // all waves done reading `cur`, writes visible
  }

  // ---- epilogue: C-frag mapping row = j + 8*half, col = l16 ----
#pragma unroll
  for (int mi = 0; mi < 2; ++mi) {
#pragma unroll
    for (int ni = 0; ni < 4; ++ni) {
      int col = n0 + wn * 64 + ni * 16 + l16;
      float bv = bias[col];
#pragma unroll
      for (int j = 0; j < 8; ++j) {
        int row = m0 + wm * 32 + mi * 16 + j + 8 * half;
        float v = (acc[mi][ni][j] + bv) * scale;
        if (GELU) v = 0.5f * v * (1.0f + erff(v * 0.70710678118654752f));
        size_t idx = (size_t)row * N + col;
        if (OUT_BF16) outB[idx] = (bf16)v;
        else          outF[idx] = v;
      }
    }
  }
}

// ---------------------------------------------------------------------------
// Flash attention: grid (S/64, H, B), block = 128 threads = 4 waves.
// Each wave owns a 16-query band; K/V tiles of 32 keys staged in LDS.
// ---------------------------------------------------------------------------
__global__ __launch_bounds__(128)
void flash_attn(const bf16* __restrict__ Q, const bf16* __restrict__ Kb,
                const bf16* __restrict__ Vb, const int* __restrict__ mask,
                bf16* __restrict__ Ctx) {
  __shared__ bf16 Ks[32][72];     // 144B row stride (16B aligned chunks)
  __shared__ bf16 Vs[32][72];
  __shared__ float Madd[32];
  __shared__ bf16 Ps[4][16][40];  // per-wave P tile scratch

  const int tid  = threadIdx.x;
  const int lane = tid & 31;
  const int wave = tid >> 5;
  const int half = lane >> 4;
  const int l16  = lane & 15;
  const int qt   = blockIdx.x;
  const int h    = blockIdx.y;
  const int b    = blockIdx.z;

  // Load this wave's 16x64 Q band as two A-fragments (dh 0..31, 32..63).
  const size_t qrow = ((size_t)b * SS + qt * 64 + wave * 16 + l16) * DD + h * DH;
  v16bf qa0 = frag_from(&Q[qrow + 0  + 8 * half]);
  v16bf qa1 = frag_from(&Q[qrow + 32 + 8 * half]);

  float m[8], l[8];
  v8f cacc[4] = {};
#pragma unroll
  for (int j = 0; j < 8; ++j) { m[j] = -1e30f; l[j] = 0.0f; }

  for (int kt = 0; kt < SS / 32; ++kt) {
    // ---- cooperative K/V tile load: 32 keys x 64 dh each ----
#pragma unroll
    for (int c = 0; c < 2; ++c) {
      int ch = tid + c * 128;                 // 0..255
      int r = ch >> 3, col = (ch & 7) * 8;
      size_t g = ((size_t)b * SS + kt * 32 + r) * DD + h * DH + col;
      *(v8bf*)&Ks[r][col] = *(const v8bf*)&Kb[g];
      *(v8bf*)&Vs[r][col] = *(const v8bf*)&Vb[g];
    }
    if (tid < 32)
      Madd[tid] = mask[(size_t)b * SS + kt * 32 + tid] ? 0.0f : -1e30f;
    __syncthreads();

    // ---- scores: S = Q x K^T  (two 16x16 key groups, K-dim = dh 64) ----
    v8f s0 = {}, s1 = {};
    {
      v16bf kb0 = frag_from(&Ks[l16][0 + 8 * half]);
      s0 = wmma_bf16(qa0, kb0, s0);
      v16bf kb1 = frag_from(&Ks[l16][32 + 8 * half]);
      s0 = wmma_bf16(qa1, kb1, s0);
      v16bf kb2 = frag_from(&Ks[16 + l16][0 + 8 * half]);
      s1 = wmma_bf16(qa0, kb2, s1);
      v16bf kb3 = frag_from(&Ks[16 + l16][32 + 8 * half]);
      s1 = wmma_bf16(qa1, kb3, s1);
    }

    // ---- online softmax over this 32-key slab ----
    const float mk0 = Madd[l16], mk1 = Madd[16 + l16];
#pragma unroll
    for (int j = 0; j < 8; ++j) {
      float a0 = s0[j] + mk0;
      float a1 = s1[j] + mk1;
      float mx = fmaxf(a0, a1);
      mx = fmaxf(mx, __shfl_xor(mx, 1));
      mx = fmaxf(mx, __shfl_xor(mx, 2));
      mx = fmaxf(mx, __shfl_xor(mx, 4));
      mx = fmaxf(mx, __shfl_xor(mx, 8));      // row max within 16-lane group
      float mn  = fmaxf(m[j], mx);
      float cor = __expf(m[j] - mn);
      float p0  = __expf(a0 - mn);
      float p1  = __expf(a1 - mn);
      float rs  = p0 + p1;
      rs += __shfl_xor(rs, 1);
      rs += __shfl_xor(rs, 2);
      rs += __shfl_xor(rs, 4);
      rs += __shfl_xor(rs, 8);
      l[j] = l[j] * cor + rs;
      m[j] = mn;
#pragma unroll
      for (int f = 0; f < 4; ++f) cacc[f][j] *= cor;
      // C-frag element of VGPR j lives at row (j + 8*half), col l16 / 16+l16
      Ps[wave][j + 8 * half][l16]      = (bf16)p0;
      Ps[wave][j + 8 * half][16 + l16] = (bf16)p1;
    }
    asm volatile("" ::: "memory");  // keep DS stores before DS reads

    // ---- ctx += P(16x32) x V(32x64) ----
    v16bf pa = frag_from(&Ps[wave][l16][8 * half]);
#pragma unroll
    for (int f = 0; f < 4; ++f) {
      v16bf vb = frag_strided(&Vs[0][f * 16 + l16], 72, half);
      cacc[f] = wmma_bf16(pa, vb, cacc[f]);
    }
    __syncthreads();  // Ks/Vs reused next iteration
  }

  // ---- finalize: divide by row sums, write bf16 ctx (b,s,h*64+dh) ----
#pragma unroll
  for (int j = 0; j < 8; ++j) {
    float inv = (l[j] > 0.0f) ? (1.0f / l[j]) : 0.0f;
    size_t o = ((size_t)b * SS + qt * 64 + wave * 16 + j + 8 * half) * DD + h * DH;
#pragma unroll
    for (int f = 0; f < 4; ++f)
      Ctx[o + f * 16 + l16] = (bf16)(cacc[f][j] * inv);
  }
}

// ---------------------------------------------------------------------------
// Residual + LayerNorm over D=1024; optionally emits bf16 copy of the result.
// One block (256 thr) per token row. outF may alias y (in-place).
// ---------------------------------------------------------------------------
__device__ __forceinline__ float block_sum(float v, float* red) {
  v += __shfl_xor(v, 1);
  v += __shfl_xor(v, 2);
  v += __shfl_xor(v, 4);
  v += __shfl_xor(v, 8);
  v += __shfl_xor(v, 16);
  int lane = threadIdx.x & 31, w = threadIdx.x >> 5;
  if (lane == 0) red[w] = v;
  __syncthreads();
  if (threadIdx.x == 0) {
    float s = 0.0f;
#pragma unroll
    for (int i = 0; i < 8; ++i) s += red[i];
    red[0] = s;
  }
  __syncthreads();
  float r = red[0];
  __syncthreads();
  return r;
}

__global__ __launch_bounds__(256)
void ln_residual(const float* y, const float* __restrict__ res,
                 const float* __restrict__ g, const float* __restrict__ beta,
                 float* outF, bf16* outB) {
  __shared__ float red[8];
  const size_t row = blockIdx.x;
  const float* yp = y + row * DD;
  const float* rp = res + row * DD;
  const int base = threadIdx.x * 4;

  float v[4];
  float s = 0.0f;
#pragma unroll
  for (int i = 0; i < 4; ++i) { v[i] = yp[base + i] + rp[base + i]; s += v[i]; }
  float mu = block_sum(s, red) * (1.0f / DD);

  float q = 0.0f;
#pragma unroll
  for (int i = 0; i < 4; ++i) { float d = v[i] - mu; q += d * d; }
  float var = block_sum(q, red) * (1.0f / DD);
  float rstd = rsqrtf(var + 1e-12f);

#pragma unroll
  for (int i = 0; i < 4; ++i) {
    float o = (v[i] - mu) * rstd * g[base + i] + beta[base + i];
    outF[row * DD + base + i] = o;
    if (outB) outB[row * DD + base + i] = (bf16)o;
  }
}

// ---------------------------------------------------------------------------
// Launch
// ---------------------------------------------------------------------------
extern "C" void kernel_launch(void* const* d_in, const int* in_sizes, int n_in,
                              void* d_out, int out_size, void* d_ws, size_t ws_size,
                              hipStream_t stream) {
  (void)in_sizes; (void)n_in; (void)out_size; (void)ws_size;

  const float* x    = (const float*)d_in[0];
  const int*   msk  = (const int*)d_in[1];
  const float* Wq   = (const float*)d_in[2];
  const float* bq   = (const float*)d_in[3];
  const float* Wk   = (const float*)d_in[4];
  const float* bk   = (const float*)d_in[5];
  const float* Wv   = (const float*)d_in[6];
  const float* bv   = (const float*)d_in[7];
  const float* Wo   = (const float*)d_in[8];
  const float* bo   = (const float*)d_in[9];
  const float* ln1g = (const float*)d_in[10];
  const float* ln1b = (const float*)d_in[11];
  const float* W1   = (const float*)d_in[12];
  const float* b1   = (const float*)d_in[13];
  const float* W2   = (const float*)d_in[14];
  const float* b2   = (const float*)d_in[15];
  const float* ln2g = (const float*)d_in[16];
  const float* ln2b = (const float*)d_in[17];
  float* out = (float*)d_out;

  char* ws = (char*)d_ws;
  const size_t MB = 1ull << 20;
  bf16*  wqb  = (bf16*)(ws + 0 * MB);
  bf16*  wkb  = (bf16*)(ws + 2 * MB);
  bf16*  wvb  = (bf16*)(ws + 4 * MB);
  bf16*  wob  = (bf16*)(ws + 6 * MB);
  bf16*  w1b  = (bf16*)(ws + 8 * MB);     // F*D
  bf16*  w2b  = (bf16*)(ws + 16 * MB);    // D*F
  bf16*  xb   = (bf16*)(ws + 24 * MB);
  bf16*  qb   = (bf16*)(ws + 40 * MB);
  bf16*  kb   = (bf16*)(ws + 56 * MB);
  bf16*  vb   = (bf16*)(ws + 72 * MB);
  bf16*  ctxb = (bf16*)(ws + 88 * MB);
  float* sa   = (float*)(ws + 104 * MB);  // sa_out, then h (in-place LN1)
  bf16*  hb   = (bf16*)(ws + 136 * MB);
  bf16*  ffb  = (bf16*)(ws + 40 * MB);    // reuse q/k/v/ctx region (dead by FFN1)
  float* y2   = (float*)(ws + 152 * MB);

  const size_t nD = (size_t)DD * DD, nF = (size_t)FF * DD, nX = (size_t)NTOK * DD;
  dim3 cvtG(2048), cvtB(256);
  hipLaunchKernelGGL(cvt_f32_bf16, cvtG, cvtB, 0, stream, x,  xb,  nX);
  hipLaunchKernelGGL(cvt_f32_bf16, cvtG, cvtB, 0, stream, Wq, wqb, nD);
  hipLaunchKernelGGL(cvt_f32_bf16, cvtG, cvtB, 0, stream, Wk, wkb, nD);
  hipLaunchKernelGGL(cvt_f32_bf16, cvtG, cvtB, 0, stream, Wv, wvb, nD);
  hipLaunchKernelGGL(cvt_f32_bf16, cvtG, cvtB, 0, stream, Wo, wob, nD);
  hipLaunchKernelGGL(cvt_f32_bf16, cvtG, cvtB, 0, stream, W1, w1b, nF);
  hipLaunchKernelGGL(cvt_f32_bf16, cvtG, cvtB, 0, stream, W2, w2b, nF);

  dim3 gB(256);
  dim3 gQKV(NTOK / 128, DD / 128);
  // q = (x Wq^T + bq) / sqrt(64)
  hipLaunchKernelGGL((gemm_bf16_nt<false, true>), gQKV, gB, 0, stream,
                     xb, wqb, bq, 0.125f, (float*)nullptr, qb, NTOK, DD, DD);
  hipLaunchKernelGGL((gemm_bf16_nt<false, true>), gQKV, gB, 0, stream,
                     xb, wkb, bk, 1.0f, (float*)nullptr, kb, NTOK, DD, DD);
  hipLaunchKernelGGL((gemm_bf16_nt<false, true>), gQKV, gB, 0, stream,
                     xb, wvb, bv, 1.0f, (float*)nullptr, vb, NTOK, DD, DD);

  dim3 aG(SS / 64, HH, BB), aB(128);
  hipLaunchKernelGGL(flash_attn, aG, aB, 0, stream, qb, kb, vb, msk, ctxb);

  // sa = ctx Wo^T + bo   (f32)
  hipLaunchKernelGGL((gemm_bf16_nt<false, false>), gQKV, gB, 0, stream,
                     ctxb, wob, bo, 1.0f, sa, (bf16*)nullptr, NTOK, DD, DD);

  // h = LN1(sa + x), in-place in sa; also bf16 copy hb
  hipLaunchKernelGGL(ln_residual, dim3(NTOK), dim3(256), 0, stream,
                     sa, x, ln1g, ln1b, sa, hb);

  // ff = gelu(h W1^T + b1)   (bf16, M=8192 N=4096 K=1024)
  dim3 gF1(NTOK / 128, FF / 128);
  hipLaunchKernelGGL((gemm_bf16_nt<true, true>), gF1, gB, 0, stream,
                     hb, w1b, b1, 1.0f, (float*)nullptr, ffb, NTOK, FF, DD);

  // y2 = ff W2^T + b2   (f32, K=4096)
  hipLaunchKernelGGL((gemm_bf16_nt<false, false>), gQKV, gB, 0, stream,
                     ffb, w2b, b2, 1.0f, y2, (bf16*)nullptr, NTOK, DD, FF);

  // out = LN2(y2 + h)
  hipLaunchKernelGGL(ln_residual, dim3(NTOK), dim3(256), 0, stream,
                     y2, sa, ln2g, ln2b, out, (bf16*)nullptr);
}
